// Net_33191507263722
// MI455X (gfx1250) — matrix operand order repaired and checked by reading, not compiled, not run on hardware
//
#include <hip/hip_runtime.h>
#include <math.h>
#include <stdint.h>

#define N_NODES  40000
#define N_EDGES  640000
#define HIDDEN   128
#define N_LAYERS 4
#define N_CLASSES 40

typedef __attribute__((ext_vector_type(16))) __bf16 v16bf;
typedef __attribute__((ext_vector_type(8)))  __bf16 v8bf;
typedef __attribute__((ext_vector_type(8)))  float  v8f;

// ---------------------------------------------------------------------------
// CSR row_ptr from sorted dst (edge-parallel boundary fill)
// ---------------------------------------------------------------------------
__global__ void build_row_ptr(const int* __restrict__ dst, int* __restrict__ row_ptr) {
    int e = blockIdx.x * blockDim.x + threadIdx.x;
    if (e >= N_EDGES) return;
    int d  = dst[e];
    int dp = (e == 0) ? -1 : dst[e - 1];
    for (int n = dp + 1; n <= d; ++n) row_ptr[n] = e;
    if (e == N_EDGES - 1) {
        for (int n = d + 1; n <= N_NODES; ++n) row_ptr[n] = N_EDGES;
    }
}

// ---------------------------------------------------------------------------
// f32 -> bf16 conversion of activations
// ---------------------------------------------------------------------------
__global__ void conv_h(const float* __restrict__ h, __bf16* __restrict__ hbf) {
    int i = blockIdx.x * blockDim.x + threadIdx.x;
    if (i < N_NODES * HIDDEN) hbf[i] = (__bf16)h[i];
}

// Weights for layer l -> bf16, transposed to [mat][n][k] so a B-fragment lane
// reads 16 contiguous bf16 (32B): lanes 0-15 hold K=base..base+15 of column
// N=lane&15, lanes 16-31 hold K=base+16..base+31 (contiguous k order, same as the A fragment).
__global__ void conv_w(const float* __restrict__ Wq, const float* __restrict__ Wk,
                       const float* __restrict__ Wv, __bf16* __restrict__ WbfT, int l) {
    int i = blockIdx.x * blockDim.x + threadIdx.x;           // m*16384 + n*128 + k
    if (i >= 3 * HIDDEN * HIDDEN) return;
    int m   = i >> 14;
    int rem = i & 16383;
    int n   = rem >> 7;
    int kk  = rem & 127;
    const float* W = (m == 0 ? Wq : (m == 1 ? Wk : Wv)) + l * HIDDEN * HIDDEN;
    WbfT[i] = (__bf16)W[(kk << 7) + n];                      // W is [k][n] row-major
}

// ---------------------------------------------------------------------------
// Helper: async global->LDS 16B copy (CDNA5, ASYNCcnt-tracked)
// ---------------------------------------------------------------------------
__device__ __forceinline__ void async_cp16(unsigned lds_off, const void* gptr) {
    asm volatile("global_load_async_to_lds_b128 %0, %1, off"
                 :: "v"(lds_off), "v"((unsigned long long)(uintptr_t)gptr)
                 : "memory");
}
__device__ __forceinline__ void async_wait0() {
    asm volatile("s_wait_asynccnt 0x0" ::: "memory");
}

// ---------------------------------------------------------------------------
// Fused Q/K/V projection.
// grid = (ceil(2500/8), 3), block = 256 (8 waves).
// Block stages W (32KB) + a 128x128 h tile (32KB) into LDS via async copies;
// each wave computes a 16-row x 128-col strip: 8 f32 accumulators, 32 WMMAs.
// ---------------------------------------------------------------------------
__global__ void qkv_wmma(const __bf16* __restrict__ hbf, const __bf16* __restrict__ WbfT,
                         const float* __restrict__ bq, const float* __restrict__ bk,
                         const float* __restrict__ bv,
                         float* __restrict__ q, float* __restrict__ k, float* __restrict__ v,
                         float scaling) {
    __shared__ __bf16 sW[HIDDEN * HIDDEN];   // 32 KB, [n][k]
    __shared__ __bf16 sA[128 * HIDDEN];      // 32 KB, [row][k]

    const int sel     = blockIdx.y;          // 0:q 1:k 2:v
    const int tid     = threadIdx.x;
    const int wave    = tid >> 5;
    const int lane    = tid & 31;
    const int half    = lane >> 4;
    const int l15     = lane & 15;
    const int rowBase = blockIdx.x * 128;

    const __bf16* Wt = WbfT + sel * HIDDEN * HIDDEN;
    const unsigned sWoff = (unsigned)(uintptr_t)(void*)sW;
    const unsigned sAoff = (unsigned)(uintptr_t)(void*)sA;

    // ---- async stage: 2048 x 16B chunks each for W and A (8 iters/thread) ----
#pragma unroll
    for (int it = 0; it < 8; ++it) {
        int li = it * 256 + tid;                       // chunk index, 16B chunks
        // weights: linear
        async_cp16(sWoff + li * 16, (const char*)Wt + li * 16);
        // h tile: clamp row for the ragged last block (compute is guarded)
        int row  = li >> 4;                            // 8 elems/chunk,16 chunks/row
        int kofs = (li & 15) * 8;
        int grow = rowBase + row;
        if (grow > N_NODES - 1) grow = N_NODES - 1;
        async_cp16(sAoff + li * 16, hbf + grow * HIDDEN + kofs);
    }
    async_wait0();
    __syncthreads();

    const int rowTile = rowBase + wave * 16;
    if (rowTile < N_NODES) {
        const float* bias = (sel == 0) ? bq : (sel == 1) ? bk : bv;
        float*       outp = (sel == 0) ? q  : (sel == 1) ? k  : v;
        const float  sc   = (sel == 0) ? scaling : 1.0f;

        v8f acc[8];
#pragma unroll
        for (int c = 0; c < 8; ++c) {
            float bcol = bias[c * 16 + l15];
#pragma unroll
            for (int r = 0; r < 8; ++r) acc[c][r] = bcol;
        }

#pragma unroll
        for (int kc = 0; kc < HIDDEN / 32; ++kc) {
            // A fragment from LDS (contiguous K=half*16..half*16+15, matches B)
            const __bf16* ap = &sA[(wave * 16 + l15) * HIDDEN + kc * 32 + half * 16];
            v8bf alo = *(const v8bf*)(ap);
            v8bf ahi = *(const v8bf*)(ap + 8);
            v16bf a  = __builtin_shufflevector(alo, ahi,
                        0,1,2,3,4,5,6,7,8,9,10,11,12,13,14,15);
#pragma unroll
            for (int c = 0; c < 8; ++c) {
                // B fragment: 16 contiguous bf16 of column c*16 + l15
                v16bf b = *(const v16bf*)(&sW[(c * 16 + l15) * HIDDEN + kc * 32 + half * 16]);
                acc[c] = __builtin_amdgcn_wmma_f32_16x16x32_bf16(
                           false, a, false, b, (short)0, acc[c], false, false);
            }
        }

        // D layout: lanes 0-15 -> M=r, lanes 16-31 -> M=r+8; N = lane&15
#pragma unroll
        for (int c = 0; c < 8; ++c) {
#pragma unroll
            for (int r = 0; r < 8; ++r) {
                int row = rowTile + half * 8 + r;
                outp[row * HIDDEN + c * 16 + l15] = acc[c][r] * sc;
            }
        }
    }
}

// ---------------------------------------------------------------------------
// SDDMM: score_e = <q[dst_e], k[src_e]>, one wave per edge
// ---------------------------------------------------------------------------
__global__ void sddmm(const float* __restrict__ q, const float* __restrict__ kmat,
                      const int* __restrict__ src, const int* __restrict__ dst,
                      float* __restrict__ score) {
    int gw = (int)((blockIdx.x * blockDim.x + threadIdx.x) >> 5);
    if (gw >= N_EDGES) return;
    int lane = threadIdx.x & 31;
    int d = dst[gw], s = src[gw];
    float4 qa = *(const float4*)(q    + d * HIDDEN + lane * 4);
    float4 ka = *(const float4*)(kmat + s * HIDDEN + lane * 4);
    float p = qa.x * ka.x + qa.y * ka.y + qa.z * ka.z + qa.w * ka.w;
#pragma unroll
    for (int off = 16; off > 0; off >>= 1) p += __shfl_xor(p, off, 32);
    if (lane == 0) score[gw] = p;
}

// ---------------------------------------------------------------------------
// Edge softmax + SpMM: one wave per destination node over its CSR segment
// ---------------------------------------------------------------------------
__global__ void softmax_spmm(const float* __restrict__ score, const int* __restrict__ src,
                             const int* __restrict__ row_ptr, const float* __restrict__ v,
                             float* __restrict__ hout) {
    int node = blockIdx.x * (blockDim.x >> 5) + (threadIdx.x >> 5);
    if (node >= N_NODES) return;
    int lane = threadIdx.x & 31;
    int e0 = row_ptr[node], e1 = row_ptr[node + 1];

    float m = -3.402823466e38f;
    for (int e = e0 + lane; e < e1; e += 32) m = fmaxf(m, score[e]);
#pragma unroll
    for (int off = 16; off > 0; off >>= 1) m = fmaxf(m, __shfl_xor(m, off, 32));

    float ssum = 0.0f;
    for (int e = e0 + lane; e < e1; e += 32) ssum += __expf(score[e] - m);
#pragma unroll
    for (int off = 16; off > 0; off >>= 1) ssum += __shfl_xor(ssum, off, 32);
    float inv = 1.0f / fmaxf(ssum, 1e-20f);

    float4 acc = make_float4(0.f, 0.f, 0.f, 0.f);
    for (int e = e0; e < e1; ++e) {
        if (e + 1 < e1)  // overlap next gathered v-row with current FMAs
            __builtin_prefetch(v + src[e + 1] * HIDDEN + lane * 4, 0, 0);
        float a = __expf(score[e] - m) * inv;
        int sr = src[e];
        float4 vv = *(const float4*)(v + sr * HIDDEN + lane * 4);
        acc.x += a * vv.x; acc.y += a * vv.y; acc.z += a * vv.z; acc.w += a * vv.w;
    }
    *(float4*)(hout + node * HIDDEN + lane * 4) = acc;
}

// ---------------------------------------------------------------------------
// Output projection + log_softmax, one wave per node (40 classes = 32 + 8)
// ---------------------------------------------------------------------------
__global__ void out_proj(const float* __restrict__ h, const float* __restrict__ Wout,
                         const float* __restrict__ bout, float* __restrict__ out) {
    int node = blockIdx.x * (blockDim.x >> 5) + (threadIdx.x >> 5);
    if (node >= N_NODES) return;
    int lane = threadIdx.x & 31;

    float acc0 = bout[lane];
    float acc1 = (lane < 8) ? bout[32 + lane] : 0.0f;
    const float* hr = h + node * HIDDEN;
    for (int kk = 0; kk < HIDDEN; ++kk) {
        float hv = hr[kk];
        acc0 += hv * Wout[kk * N_CLASSES + lane];
        if (lane < 8) acc1 += hv * Wout[kk * N_CLASSES + 32 + lane];
    }
    float m = (lane < 8) ? fmaxf(acc0, acc1) : acc0;
#pragma unroll
    for (int off = 16; off > 0; off >>= 1) m = fmaxf(m, __shfl_xor(m, off, 32));
    float s = __expf(acc0 - m) + ((lane < 8) ? __expf(acc1 - m) : 0.0f);
#pragma unroll
    for (int off = 16; off > 0; off >>= 1) s += __shfl_xor(s, off, 32);
    float lse = m + __logf(s);

    out[node * N_CLASSES + lane] = acc0 - lse;
    if (lane < 8) out[node * N_CLASSES + 32 + lane] = acc1 - lse;
}

// ---------------------------------------------------------------------------
extern "C" void kernel_launch(void* const* d_in, const int* in_sizes, int n_in,
                              void* d_out, int out_size, void* d_ws, size_t ws_size,
                              hipStream_t stream) {
    const float* h_in = (const float*)d_in[0];
    const int*   src  = (const int*)  d_in[1];
    const int*   dst  = (const int*)  d_in[2];
    const float* Wq   = (const float*)d_in[3];
    const float* bq   = (const float*)d_in[4];
    const float* Wk   = (const float*)d_in[5];
    const float* bk   = (const float*)d_in[6];
    const float* Wv   = (const float*)d_in[7];
    const float* bv   = (const float*)d_in[8];
    const float* Wout = (const float*)d_in[9];
    const float* bout = (const float*)d_in[10];

    const float scaling = 1.0f / sqrtf((float)HIDDEN);

    // workspace layout (256B aligned slices)
    char*  w = (char*)d_ws;
    size_t o = 0;
    auto alloc = [&](size_t bytes) -> void* {
        void* p = w + o;
        o = (o + bytes + 255) & ~(size_t)255;
        return p;
    };
    float*  hA     = (float*) alloc((size_t)N_NODES * HIDDEN * 4);
    float*  hB     = (float*) alloc((size_t)N_NODES * HIDDEN * 4);
    __bf16* hbf    = (__bf16*)alloc((size_t)N_NODES * HIDDEN * 2);
    __bf16* WbfT   = (__bf16*)alloc((size_t)3 * HIDDEN * HIDDEN * 2);
    float*  qbuf   = (float*) alloc((size_t)N_NODES * HIDDEN * 4);
    float*  kbuf   = (float*) alloc((size_t)N_NODES * HIDDEN * 4);
    float*  vbuf   = (float*) alloc((size_t)N_NODES * HIDDEN * 4);
    float*  score  = (float*) alloc((size_t)N_EDGES * 4);
    int*    rowptr = (int*)   alloc((size_t)(N_NODES + 1) * 4);

    build_row_ptr<<<(N_EDGES + 255) / 256, 256, 0, stream>>>(dst, rowptr);

    const float* cur = h_in;
    float* ping = hA;
    float* pong = hB;

    const int nRowTiles16 = N_NODES / 16;              // 2500
    const int gx = (nRowTiles16 + 7) / 8;              // 313 blocks of 128 rows

    for (int l = 0; l < N_LAYERS; ++l) {
        conv_h<<<(N_NODES * HIDDEN + 255) / 256, 256, 0, stream>>>(cur, hbf);
        conv_w<<<(3 * HIDDEN * HIDDEN + 255) / 256, 256, 0, stream>>>(Wq, Wk, Wv, WbfT, l);

        dim3 g(gx, 3);
        qkv_wmma<<<g, 256, 0, stream>>>(hbf, WbfT,
                                        bq + l * HIDDEN, bk + l * HIDDEN, bv + l * HIDDEN,
                                        qbuf, kbuf, vbuf, scaling);

        sddmm<<<(N_EDGES * 32 + 255) / 256, 256, 0, stream>>>(qbuf, kbuf, src, dst, score);

        softmax_spmm<<<(N_NODES + 7) / 8, 256, 0, stream>>>(score, src, rowptr, vbuf, ping);

        cur = ping;
        float* t = ping; ping = pong; pong = t;
    }

    out_proj<<<(N_NODES + 7) / 8, 256, 0, stream>>>(cur, Wout, bout, (float*)d_out);
}
